// NeuralODE_69965017252106
// MI455X (gfx1250) — matrix operand order repaired and checked by reading, not compiled, not run on hardware
//
#include <hip/hip_runtime.h>
#include <math.h>

// Problem constants (from reference)
#define BATCH   1024
#define DDIM    64
#define FDIM    8
#define HDIM    256
#define TEVAL   50
#define TUPTS   128
#define NSUB    4
#define NSTEPS  ((TEVAL - 1) * NSUB)   // 196
#define KZ      96                     // 72 padded to 3*32 for f16 WMMA K
#define MROWS   16                     // batch rows per workgroup
#define NWG     (BATCH / MROWS)        // 64 workgroups

typedef __attribute__((ext_vector_type(16))) _Float16 v16h;
typedef __attribute__((ext_vector_type(8)))  float    v8f;

// ---- LDS byte offsets (all 16B aligned) ----
#define OFF_W1T 0        // f16 [256][96]   49152 B  (W1^T, K padded; init staging)
#define OFF_W2T 49152    // f16 [64][256]   32768 B  (W2^T; init staging)
#define OFF_HSH 81920    // f16 [16][256]    8192 B  (hidden activations)
#define OFF_ZSH 90112    // f16 [16][96]     3072 B  (input [x|u] padded)
#define OFF_XSH 93184    // f32 [16][64]     4096 B  (state)
#define OFF_KSH 97280    // f32 [6][16][64] 24576 B  (RK stage outputs)
#define OFF_B1S 121856   // f32 [256]
#define OFF_B2S 122880   // f32 [64]
#define SMEM_BYTES 123136

// A-matrix fragment (16x32 f16): lane holds row m=lane%16; K runs
// [kh, kh+16), kh = (lane/16)*16  -> two 16B LDS loads.
static __device__ inline v16h lds_fragA(const _Float16* base, int f16off) {
    union { v16h h; uint4 q[2]; } u;
    u.q[0] = *reinterpret_cast<const uint4*>(base + f16off);
    u.q[1] = *reinterpret_cast<const uint4*>(base + f16off + 8);
    return u.h;
}

// B-matrix fragment (32x16 f16): lane holds column n=lane%16;
// lanes 0-15: K=0..15, lanes 16-31: K=16..31 -> 32 contiguous bytes
// when the weight matrix is stored [N][K] in LDS.
static __device__ inline v16h lds_fragB(const _Float16* base, int f16off) {
    union { v16h h; uint4 q[2]; } u;
    u.q[0] = *reinterpret_cast<const uint4*>(base + f16off);
    u.q[1] = *reinterpret_cast<const uint4*>(base + f16off + 8);
    return u.h;
}

static __device__ inline v8f wmma_f16(v16h a, v16h b, v8f c) {
    return __builtin_amdgcn_wmma_f32_16x16x32_f16(
        /*neg_a=*/false, a, /*neg_b=*/false, b,
        /*c_mod=*/(short)0, c, /*reuse_a=*/false, /*reuse_b=*/false);
}

// Hardware tanh on gfx1250 (V_TANH_F32); branch-free fallback otherwise.
static __device__ inline float fast_tanh(float x) {
#if __has_builtin(__builtin_amdgcn_tanhf)
    return __builtin_amdgcn_tanhf(x);
#else
    float ax = __builtin_fabsf(x);
    float e  = __builtin_amdgcn_exp2f(-2.885390081777927f * ax);
    float r  = (1.0f - e) * __builtin_amdgcn_rcpf(1.0f + e);
    return __builtin_copysignf(r, x);
#endif
}

__global__ __launch_bounds__(256, 1)
void neural_ode_dopri5_persistent(const float* __restrict__ x0,
                                  const float* __restrict__ u_batch,
                                  const float* __restrict__ W1,
                                  const float* __restrict__ b1,
                                  const float* __restrict__ W2,
                                  const float* __restrict__ b2,
                                  float* __restrict__ out) {
    extern __shared__ char smem[];
    _Float16* w1t = reinterpret_cast<_Float16*>(smem + OFF_W1T);
    _Float16* w2t = reinterpret_cast<_Float16*>(smem + OFF_W2T);
    _Float16* hsh = reinterpret_cast<_Float16*>(smem + OFF_HSH);
    _Float16* zsh = reinterpret_cast<_Float16*>(smem + OFF_ZSH);
    float*    xsh = reinterpret_cast<float*>(smem + OFF_XSH);
    float*    ksh = reinterpret_cast<float*>(smem + OFF_KSH);
    float*    b1s = reinterpret_cast<float*>(smem + OFF_B1S);
    float*    b2s = reinterpret_cast<float*>(smem + OFF_B2S);

    const int tid  = threadIdx.x;
    const int wg   = blockIdx.x;
    const int lane = tid & 31;
    const int wave = tid >> 5;
    const int mrow = lane & 15;
    const int mhalf = (lane >> 4) << 3;   // 0 or 8 : M offset for C/D rows
    const int khA   = (lane >> 4) << 4;   // 0 or 16: K-half base for A frags
    const int khB   = (lane >> 4) << 4;   // 0 or 16: K-half base for B frags

    // ---- one-time init: weights -> LDS (f16, transposed), biases, state ----
    for (int i = tid; i < HDIM * KZ; i += 256) {      // w1t[n][k]
        int n = i / KZ, k = i % KZ;
        w1t[i] = (k < (DDIM + FDIM)) ? (_Float16)W1[k * HDIM + n] : (_Float16)0.0f;
    }
    for (int i = tid; i < DDIM * HDIM; i += 256) {    // w2t[n][k]
        int n = i >> 8, k = i & 255;
        w2t[i] = (_Float16)W2[k * DDIM + n];
    }
    for (int i = tid; i < HDIM; i += 256) b1s[i] = b1[i];
    for (int i = tid; i < DDIM; i += 256) b2s[i] = b2[i];
    for (int i = tid; i < MROWS * KZ; i += 256) zsh[i] = (_Float16)0.0f;
    for (int i = tid; i < MROWS * DDIM; i += 256) {
        xsh[i] = x0[wg * (MROWS * DDIM) + i];
        int m = i >> 6, d = i & 63;
        out[(size_t)(wg * MROWS + m) * (TEVAL * DDIM) + d] = xsh[i]; // t_eval[0]
    }
    __syncthreads();

    // ---- hoist loop-invariant weight B-fragments + biases into registers ----
    v16h bw1[2][3];
    float bias1[2];
    #pragma unroll
    for (int tt = 0; tt < 2; ++tt) {
        int ncol = (wave + tt * 8) * 16 + mrow;
        #pragma unroll
        for (int c = 0; c < 3; ++c)
            bw1[tt][c] = lds_fragB(w1t, ncol * KZ + c * 32 + khB);
        bias1[tt] = b1s[ncol];
    }
    v16h bw2[8];
    float bias2 = 0.0f;
    const int ncol2 = wave * 16 + mrow;   // GEMM2 column (waves 0-3 only)
    if (wave < 4) {
        #pragma unroll
        for (int c = 0; c < 8; ++c)
            bw2[c] = lds_fragB(w2t, ncol2 * HDIM + c * 32 + khB);
        bias2 = b2s[ncol2];
    }

    // Dopri5 tableau (row 6 = solution weights)
    const float Ac[7][6] = {
        {0.f, 0.f, 0.f, 0.f, 0.f, 0.f},
        {1.f/5.f, 0.f, 0.f, 0.f, 0.f, 0.f},
        {3.f/40.f, 9.f/40.f, 0.f, 0.f, 0.f, 0.f},
        {44.f/45.f, -56.f/15.f, 32.f/9.f, 0.f, 0.f, 0.f},
        {19372.f/6561.f, -25360.f/2187.f, 64448.f/6561.f, -212.f/729.f, 0.f, 0.f},
        {9017.f/3168.f, -355.f/33.f, 46732.f/5247.f, 49.f/176.f, -5103.f/18656.f, 0.f},
        {35.f/384.f, 0.f, 500.f/1113.f, 125.f/192.f, -2187.f/6784.f, 11.f/84.f}};
    const float Cc[6] = {0.f, 0.2f, 0.3f, 0.8f, 8.f/9.f, 1.f};
    const float dt = 1.0f / (float)NSTEPS;

    for (int st = 0; st < NSTEPS; ++st) {
        const float t = (float)st * dt;

        #pragma unroll
        for (int s = 0; s < 6; ++s) {
            const float ts = t + Cc[s] * dt;

            // ---- build z = [x_stage | u(ts)] in f16 ----
            for (int i = tid; i < MROWS * DDIM; i += 256) {
                float acc = xsh[i];
                #pragma unroll
                for (int j = 0; j < s; ++j)
                    acc += dt * Ac[s][j] * ksh[j * (MROWS * DDIM) + i];
                int m = i >> 6, d = i & 63;
                zsh[m * KZ + d] = (_Float16)acc;
            }
            if (tid < MROWS * FDIM) {   // interp u on uniform grid
                int m = tid >> 3, f = tid & 7;
                float tq = ts * (float)(TUPTS - 1);
                int idx = (int)floorf(tq);
                idx = idx < 0 ? 0 : (idx > TUPTS - 2 ? TUPTS - 2 : idx);
                float w = tq - (float)idx;
                const float* ub = u_batch + (size_t)(wg * MROWS + m) * (TUPTS * FDIM)
                                  + idx * FDIM + f;
                float u0 = ub[0], u1 = ub[FDIM];
                zsh[m * KZ + DDIM + f] = (_Float16)(u0 + w * (u1 - u0));
            }
            __syncthreads();

            // ---- GEMM1: (16x96)@(96x256), bias + tanh -> hsh (f16) ----
            v16h a0 = lds_fragA(zsh, mrow * KZ +  0 + khA);
            v16h a1 = lds_fragA(zsh, mrow * KZ + 32 + khA);
            v16h a2 = lds_fragA(zsh, mrow * KZ + 64 + khA);
            #pragma unroll
            for (int tt = 0; tt < 2; ++tt) {
                int ncol = (wave + tt * 8) * 16 + mrow;
                v8f acc = {};
                acc = wmma_f16(a0, bw1[tt][0], acc);
                acc = wmma_f16(a1, bw1[tt][1], acc);
                acc = wmma_f16(a2, bw1[tt][2], acc);
                #pragma unroll
                for (int r = 0; r < 8; ++r)
                    hsh[(r + mhalf) * HDIM + ncol] =
                        (_Float16)fast_tanh(acc[r] + bias1[tt]);
            }
            __syncthreads();

            // ---- GEMM2: (16x256)@(256x64) + b2 -> k_s (f32, LDS) ----
            if (wave < 4) {
                v8f acc = {};
                #pragma unroll
                for (int c = 0; c < 8; ++c) {
                    v16h a = lds_fragA(hsh, mrow * HDIM + c * 32 + khA);
                    acc = wmma_f16(a, bw2[c], acc);
                }
                #pragma unroll
                for (int r = 0; r < 8; ++r)
                    ksh[s * (MROWS * DDIM) + (r + mhalf) * DDIM + ncol2] =
                        acc[r] + bias2;
            }
            __syncthreads();
        }

        // ---- combine: x += dt * sum b_j k_j ----
        for (int i = tid; i < MROWS * DDIM; i += 256) {
            float acc = xsh[i];
            #pragma unroll
            for (int j = 0; j < 6; ++j)
                acc += dt * Ac[6][j] * ksh[j * (MROWS * DDIM) + i];
            xsh[i] = acc;
        }
        __syncthreads();

        if ((st & (NSUB - 1)) == (NSUB - 1)) {   // snapshot at coarse grid point
            int ti = (st >> 2) + 1;
            for (int i = tid; i < MROWS * DDIM; i += 256) {
                int m = i >> 6, d = i & 63;
                out[(size_t)(wg * MROWS + m) * (TEVAL * DDIM) + ti * DDIM + d] = xsh[i];
            }
            __syncthreads();
        }
    }
}

extern "C" void kernel_launch(void* const* d_in, const int* in_sizes, int n_in,
                              void* d_out, int out_size, void* d_ws, size_t ws_size,
                              hipStream_t stream) {
    (void)in_sizes; (void)n_in; (void)out_size; (void)d_ws; (void)ws_size;
    const float* x0      = (const float*)d_in[0];
    // d_in[1] = t_eval, d_in[2] = t_u : uniform grids, derived analytically
    const float* u_batch = (const float*)d_in[3];
    const float* W1      = (const float*)d_in[4];
    const float* b1      = (const float*)d_in[5];
    const float* W2      = (const float*)d_in[6];
    const float* b2      = (const float*)d_in[7];
    float* out = (float*)d_out;

    hipFuncSetAttribute(reinterpret_cast<const void*>(neural_ode_dopri5_persistent),
                        hipFuncAttributeMaxDynamicSharedMemorySize, SMEM_BYTES);
    neural_ode_dopri5_persistent<<<NWG, 256, SMEM_BYTES, stream>>>(
        x0, u_batch, W1, b1, W2, b2, out);
}
